// GATNode_14525579395555
// MI455X (gfx1250) — matrix operand order repaired and checked by reading, not compiled, not run on hardware
//
#include <hip/hip_runtime.h>
#include <math.h>

// Problem constants (from reference)
#define NN   50000     // nodes
#define NE   800000    // edges (self-loops appended -> NE+NN)
#define FIN  128
#define HIDC 64
#define NH   4
#define OUTC 40
#define DH   256       // NH*HIDC
#define MPAD 50048     // 391 * 128 (GEMM row padding)

typedef __attribute__((ext_vector_type(16))) __bf16       v16bf;
typedef __attribute__((ext_vector_type(8)))  float        v8f;
typedef __attribute__((ext_vector_type(4)))  float        v4f;
typedef __attribute__((ext_vector_type(4)))  unsigned int v4u;
typedef __attribute__((ext_vector_type(4)))  int          v4i;

// address-space-typed pointers for the async global->LDS builtin
typedef __attribute__((address_space(1))) v4i as1_v4i;
typedef __attribute__((address_space(3))) v4i as3_v4i;

static __device__ __forceinline__ unsigned short f2bf(float f) {
  unsigned int u = __float_as_uint(f);
  u += 0x7FFFu + ((u >> 16) & 1u);          // round-to-nearest-even
  return (unsigned short)(u >> 16);
}

// ---- CDNA5 async global->LDS (ASYNCcnt) helpers -----------------------------
static __device__ __forceinline__ void async_ld_b128(const void* g, void* l) {
#if __has_builtin(__builtin_amdgcn_global_load_async_to_lds_b128)
  __builtin_amdgcn_global_load_async_to_lds_b128(
      (as1_v4i*)(unsigned long long)g,
      (as3_v4i*)(unsigned)(unsigned long long)l, 0, 0);
#else
  unsigned lo = (unsigned)(unsigned long long)l;   // LDS byte offset (addr[31:0])
  asm volatile("global_load_async_to_lds_b128 %0, %1, off"
               :: "v"(lo), "v"(g) : "memory");
#endif
}
static __device__ __forceinline__ void wait_async0() {
#if __has_builtin(__builtin_amdgcn_s_wait_asynccnt)
  __builtin_amdgcn_s_wait_asynccnt(0);
#else
  asm volatile("s_wait_asynccnt 0x0" ::: "memory");
#endif
}

// ---------------- conversion kernels ----------------
__global__ void k_cvt_bf16(const float* __restrict__ src,
                           unsigned short* __restrict__ dst, long long n) {
  long long i = (long long)blockIdx.x * blockDim.x + threadIdx.x;
  if (i < n) dst[i] = f2bf(src[i]);
}

// fused ELU + bf16 conversion (fp32 post-ELU tensor is never needed)
__global__ void k_elu_cvt(const float* __restrict__ src,
                          unsigned short* __restrict__ dst, long long n) {
  long long i = (long long)blockIdx.x * blockDim.x + threadIdx.x;
  if (i < n) { float v = src[i]; v = v > 0.f ? v : expm1f(v); dst[i] = f2bf(v); }
}

// W [K x N] fp32 row-major -> Wt [Np x K] bf16 (transposed, zero-padded columns)
__global__ void k_cvt_wt(const float* __restrict__ W, unsigned short* __restrict__ Wt,
                         int K, int N, int Np) {
  int i = blockIdx.x * blockDim.x + threadIdx.x;
  if (i >= Np * K) return;
  int n = i / K, k = i % K;
  Wt[n * K + k] = f2bf((n < N) ? W[k * N + n] : 0.0f);
}

// ---------------- WMMA GEMM:  H[M x Np] = Xb[M x K] * Wt'[K x Np] ----------------
// Block 256 threads (8 waves) computes 128 rows x 64 cols.
// Wave w owns rows tileM+16w..+15 and all 64 cols (4 accumulators).
// B tile (64 cols x 32 k, 4KB) double-buffered in LDS via async global->LDS.
static __device__ __forceinline__ void prefetch_B(const unsigned short* __restrict__ Wt,
                                                  unsigned short* bt, int tileN,
                                                  int K, int k0, int tid) {
  int nrow = tid >> 2;           // 0..63 (column of B = row of Wt)
  int kk   = (tid & 3) * 8;      // 0,8,16,24 (8 bf16 = 16B)
  async_ld_b128(&Wt[(size_t)(tileN + nrow) * K + k0 + kk], &bt[nrow * 32 + kk]);
}

__global__ void __launch_bounds__(256)
k_gemm_wmma(const unsigned short* __restrict__ Xb,
            const unsigned short* __restrict__ Wt,   // [Np x K] (pre-transposed)
            float* __restrict__ H, int K, int Np) {
  __shared__ unsigned short bt[2][64 * 32];          // double-buffered B tile
  const int tid   = threadIdx.x;
  const int wave  = tid >> 5;
  const int lane  = tid & 31;
  const int half  = lane >> 4;                        // 0: lanes 0-15, 1: lanes 16-31
  const int l15   = lane & 15;
  const int rowB  = blockIdx.x * 128 + wave * 16;
  const int tileN = blockIdx.y * 64;

  v8f acc[4] = {};
  prefetch_B(Wt, bt[0], tileN, K, 0, tid);

  int cur = 0;
  for (int k0 = 0; k0 < K; k0 += 32, cur ^= 1) {
    wait_async0();
    __syncthreads();
    if (k0 + 32 < K) prefetch_B(Wt, bt[cur ^ 1], tileN, K, k0 + 32, tid);

    // A fragment from global (ISA 16-bit A 16x32 layout):
    // lanes 0-15: K 0-7 / 16-23; lanes 16-31: K 8-15 / 24-31
    union { v4u u[2]; v16bf v; } a;
    const unsigned short* ap = &Xb[(size_t)(rowB + l15) * K + k0 + half * 8];
    a.u[0] = *(const v4u*)ap;
    a.u[1] = *(const v4u*)(ap + 16);

#pragma unroll
    for (int q = 0; q < 4; ++q) {
      // B fragment: lane = column (q*16+l15), K 0-7 / 16-23 (lanes 0-15), 8-15 / 24-31 (lanes 16-31)
      union { v4u u[2]; v16bf v; } b;
      const unsigned short* bp = &bt[cur][(q * 16 + l15) * 32 + half * 8];
      b.u[0] = *(const v4u*)bp;
      b.u[1] = *(const v4u*)(bp + 16);
      acc[q] = __builtin_amdgcn_wmma_f32_16x16x32_bf16(false, a.v, false, b.v,
                                                       (short)0, acc[q], false, false);
    }
  }
  // C/D layout: VGPR i -> row i + 8*half, col l15
#pragma unroll
  for (int q = 0; q < 4; ++q)
#pragma unroll
    for (int i = 0; i < 8; ++i)
      H[(size_t)(rowB + i + half * 8) * Np + tileN + q * 16 + l15] = acc[q][i];
}

// ---------------- attention logits: als/ald[n,h] = <h[n,h,:], a_src/dst[h,:]> ----------------
__global__ void k_att_logits(const float* __restrict__ H, const float* __restrict__ a_s,
                             const float* __restrict__ a_d, float* __restrict__ als,
                             float* __restrict__ ald, int n, int heads, int ch, int ldh) {
  int i = blockIdx.x * blockDim.x + threadIdx.x;
  if (i >= n * heads) return;
  int node = i / heads, h = i % heads;
  const v4f* hp = (const v4f*)&H[(size_t)node * ldh + h * ch];
  const v4f* as = (const v4f*)&a_s[h * ch];
  const v4f* ad = (const v4f*)&a_d[h * ch];
  float s = 0.f, d = 0.f;
  for (int c = 0; c < ch / 4; ++c) {
    v4f v = hp[c], vs = as[c], vd = ad[c];
#pragma unroll
    for (int j = 0; j < 4; ++j) { s += v[j] * vs[j]; d += v[j] * vd[j]; }
  }
  als[i] = s; ald[i] = d;
}

// ---------------- init kernels ----------------
__global__ void k_init_ms(float* __restrict__ m, float* __restrict__ s, int n) {
  int i = blockIdx.x * blockDim.x + threadIdx.x;
  if (i < n) { m[i] = -INFINITY; s[i] = 0.f; }
}
__global__ void k_init_out(float* __restrict__ out, const float* __restrict__ b,
                           int n, int d, int ldh) {
  long long i = (long long)blockIdx.x * blockDim.x + threadIdx.x;
  if (i >= (long long)n * ldh) return;
  int j = (int)(i % ldh);
  out[i] = (j < d) ? b[j] : 0.f;
}

// ---------------- edge kernels ----------------
static __device__ __forceinline__ void edge_ep(const long long* __restrict__ ei,
                                               int e, int E, int& s, int& d) {
  if (e < E) { s = (int)ei[e]; d = (int)ei[(size_t)E + e]; }
  else       { s = d = e - E; }                       // appended self-loops
}

__global__ void k_edge_max(const long long* __restrict__ ei, const float* __restrict__ als,
                           const float* __restrict__ ald, float* __restrict__ m,
                           int E, int n, int heads) {
  long long i = (long long)blockIdx.x * blockDim.x + threadIdx.x;
  if (i >= (long long)(E + n) * heads) return;
  int e = (int)(i / heads), h = (int)(i % heads);
  int s, d; edge_ep(ei, e, E, s, d);
  float v = als[(size_t)s * heads + h] + ald[(size_t)d * heads + h];
  v = v > 0.f ? v : 0.2f * v;                         // leaky_relu(0.2)
  float* addr = &m[(size_t)d * heads + h];
  if (v >= 0.f) atomicMax((int*)addr, __float_as_int(v));
  else          atomicMin((unsigned int*)addr, __float_as_uint(v));
}

__global__ void k_edge_exp(const long long* __restrict__ ei, const float* __restrict__ als,
                           const float* __restrict__ ald, const float* __restrict__ m,
                           float* __restrict__ ssum, float* __restrict__ eexp,
                           int E, int n, int heads) {
  long long i = (long long)blockIdx.x * blockDim.x + threadIdx.x;
  if (i >= (long long)(E + n) * heads) return;
  int e = (int)(i / heads), h = (int)(i % heads);
  int s, d; edge_ep(ei, e, E, s, d);
  float v = als[(size_t)s * heads + h] + ald[(size_t)d * heads + h];
  v = v > 0.f ? v : 0.2f * v;
  float ex = __expf(v - m[(size_t)d * heads + h]);
  eexp[i] = ex;
  atomicAdd(&ssum[(size_t)d * heads + h], ex);
}

// Wave-per-edge aggregation, specialized for heads=4, ch=64 (256 channels).
// Lane l covers channels 8l..8l+7 (all in head l/8): two float4 loads + 8 atomics.
__global__ void __launch_bounds__(256)
k_edge_aggr256(const long long* __restrict__ ei, const float* __restrict__ eexp,
               const float* __restrict__ ssum, const float* __restrict__ H,
               float* __restrict__ out, int E, int n) {
  int wid  = (int)(((long long)blockIdx.x * blockDim.x + threadIdx.x) >> 5);
  int lane = threadIdx.x & 31;
  if (wid >= E + n) return;
  int s, d; edge_ep(ei, wid, E, s, d);
  int head = lane >> 3;
  float alpha = eexp[(size_t)wid * NH + head] / ssum[(size_t)d * NH + head];
  const v4f* hp = (const v4f*)&H[(size_t)s * DH + lane * 8];
  float*     op = &out[(size_t)d * DH + lane * 8];
  v4f h0 = hp[0], h1 = hp[1];
#pragma unroll
  for (int j = 0; j < 4; ++j) atomicAdd(&op[j],     alpha * h0[j]);
#pragma unroll
  for (int j = 0; j < 4; ++j) atomicAdd(&op[4 + j], alpha * h1[j]);
}

// Wave-per-edge aggregation, generic (layer 2: heads=1, ch=40)
__global__ void __launch_bounds__(256)
k_edge_aggr_w(const long long* __restrict__ ei, const float* __restrict__ eexp,
              const float* __restrict__ ssum, const float* __restrict__ H,
              float* __restrict__ out, int E, int n, int heads, int ch, int ldh) {
  int wid  = (int)(((long long)blockIdx.x * blockDim.x + threadIdx.x) >> 5);
  int lane = threadIdx.x & 31;
  if (wid >= E + n) return;
  int s, d; edge_ep(ei, wid, E, s, d);
  int dt = heads * ch;
  for (int j = lane; j < dt; j += 32) {
    int h = j / ch;
    float alpha = eexp[(size_t)wid * heads + h] / ssum[(size_t)d * heads + h];
    atomicAdd(&out[(size_t)d * ldh + j], alpha * H[(size_t)s * ldh + j]);
  }
}

// ---------------- final log-softmax ----------------
__global__ void k_logsoftmax(const float* __restrict__ src, float* __restrict__ y,
                             int n, int d, int ldh) {
  int i = blockIdx.x * blockDim.x + threadIdx.x;
  if (i >= n) return;
  const float* p = &src[(size_t)i * ldh];
  float mx = -INFINITY;
  for (int j = 0; j < d; ++j) mx = fmaxf(mx, p[j]);
  float sum = 0.f;
  for (int j = 0; j < d; ++j) sum += __expf(p[j] - mx);
  float ls = mx + __logf(sum);
  for (int j = 0; j < d; ++j) y[(size_t)i * d + j] = p[j] - ls;
}

// ---------------- launcher ----------------
static inline int cdiv(long long a, long long b) { return (int)((a + b - 1) / b); }

extern "C" void kernel_launch(void* const* d_in, const int* in_sizes, int n_in,
                              void* d_out, int out_size, void* d_ws, size_t ws_size,
                              hipStream_t stream) {
  (void)in_sizes; (void)n_in; (void)out_size; (void)ws_size;
  const float*     x   = (const float*)d_in[0];
  const long long* ei  = (const long long*)d_in[1];   // int64 edge_index [2,E]
  const float* W0 = (const float*)d_in[2], *as0 = (const float*)d_in[3],
             *ad0 = (const float*)d_in[4], *b0  = (const float*)d_in[5];
  const float* W1 = (const float*)d_in[6], *as1 = (const float*)d_in[7],
             *ad1 = (const float*)d_in[8], *b1  = (const float*)d_in[9];
  const float* W2 = (const float*)d_in[10], *as2 = (const float*)d_in[11],
             *ad2 = (const float*)d_in[12], *b2  = (const float*)d_in[13];
  float* y = (float*)d_out;

  char* ws = (char*)d_ws;
  size_t off = 0;
  auto carve = [&](size_t bytes) -> void* {
    void* p = ws + off;
    off += (bytes + 255) & ~(size_t)255;
    return p;
  };
  unsigned short* Xb = (unsigned short*)carve((size_t)MPAD * DH * 2); // bf16 layer input
  unsigned short* Wt = (unsigned short*)carve((size_t)DH * DH * 2);   // bf16 W transposed
  float* Hb  = (float*)carve((size_t)MPAD * DH * 4);                  // h = x @ W
  float* Ob  = (float*)carve((size_t)NN * DH * 4);                    // aggregated output
  float* ALs = (float*)carve((size_t)NN * NH * 4);
  float* ALd = (float*)carve((size_t)NN * NH * 4);
  float* Mx  = (float*)carve((size_t)NN * NH * 4);
  float* Sm  = (float*)carve((size_t)NN * NH * 4);
  float* Ee  = (float*)carve((size_t)(NE + NN) * NH * 4);

  const int ET = NE + NN;
  const int GMB = MPAD / 128;   // 391 GEMM row-blocks

  // ================= layer 0 (K=128, heads=4, ch=64) =================
  k_cvt_bf16<<<cdiv((long long)NN * FIN, 256), 256, 0, stream>>>(x, Xb, (long long)NN * FIN);
  k_cvt_wt  <<<cdiv((long long)DH * FIN, 256), 256, 0, stream>>>(W0, Wt, FIN, DH, DH);
  { dim3 g(GMB, DH / 64);
    k_gemm_wmma<<<g, 256, 0, stream>>>(Xb, Wt, Hb, FIN, DH); }
  k_att_logits<<<cdiv((long long)NN * NH, 256), 256, 0, stream>>>(Hb, as0, ad0, ALs, ALd, NN, NH, HIDC, DH);
  k_init_ms  <<<cdiv((long long)NN * NH, 256), 256, 0, stream>>>(Mx, Sm, NN * NH);
  k_init_out <<<cdiv((long long)NN * DH, 256), 256, 0, stream>>>(Ob, b0, NN, DH, DH);
  k_edge_max <<<cdiv((long long)ET * NH, 256), 256, 0, stream>>>(ei, ALs, ALd, Mx, NE, NN, NH);
  k_edge_exp <<<cdiv((long long)ET * NH, 256), 256, 0, stream>>>(ei, ALs, ALd, Mx, Sm, Ee, NE, NN, NH);
  k_edge_aggr256<<<cdiv((long long)ET * 32, 256), 256, 0, stream>>>(ei, Ee, Sm, Hb, Ob, NE, NN);
  k_elu_cvt  <<<cdiv((long long)NN * DH, 256), 256, 0, stream>>>(Ob, Xb, (long long)NN * DH);

  // ================= layer 1 (K=256, heads=4, ch=64) =================
  k_cvt_wt  <<<cdiv((long long)DH * DH, 256), 256, 0, stream>>>(W1, Wt, DH, DH, DH);
  { dim3 g(GMB, DH / 64);
    k_gemm_wmma<<<g, 256, 0, stream>>>(Xb, Wt, Hb, DH, DH); }
  k_att_logits<<<cdiv((long long)NN * NH, 256), 256, 0, stream>>>(Hb, as1, ad1, ALs, ALd, NN, NH, HIDC, DH);
  k_init_ms  <<<cdiv((long long)NN * NH, 256), 256, 0, stream>>>(Mx, Sm, NN * NH);
  k_init_out <<<cdiv((long long)NN * DH, 256), 256, 0, stream>>>(Ob, b1, NN, DH, DH);
  k_edge_max <<<cdiv((long long)ET * NH, 256), 256, 0, stream>>>(ei, ALs, ALd, Mx, NE, NN, NH);
  k_edge_exp <<<cdiv((long long)ET * NH, 256), 256, 0, stream>>>(ei, ALs, ALd, Mx, Sm, Ee, NE, NN, NH);
  k_edge_aggr256<<<cdiv((long long)ET * 32, 256), 256, 0, stream>>>(ei, Ee, Sm, Hb, Ob, NE, NN);
  k_elu_cvt  <<<cdiv((long long)NN * DH, 256), 256, 0, stream>>>(Ob, Xb, (long long)NN * DH);

  // ================= layer 2 (K=256, heads=1, ch=40, N padded to 64) =================
  const int NP2 = 64;
  k_cvt_wt  <<<cdiv((long long)NP2 * DH, 256), 256, 0, stream>>>(W2, Wt, DH, OUTC, NP2);
  { dim3 g(GMB, 1);
    k_gemm_wmma<<<g, 256, 0, stream>>>(Xb, Wt, Hb, DH, NP2); }
  k_att_logits<<<cdiv((long long)NN, 256), 256, 0, stream>>>(Hb, as2, ad2, ALs, ALd, NN, 1, OUTC, NP2);
  k_init_ms  <<<cdiv((long long)NN, 256), 256, 0, stream>>>(Mx, Sm, NN);
  k_init_out <<<cdiv((long long)NN * NP2, 256), 256, 0, stream>>>(Ob, b2, NN, OUTC, NP2);
  k_edge_max <<<cdiv((long long)ET, 256), 256, 0, stream>>>(ei, ALs, ALd, Mx, NE, NN, 1);
  k_edge_exp <<<cdiv((long long)ET, 256), 256, 0, stream>>>(ei, ALs, ALd, Mx, Sm, Ee, NE, NN, 1);
  k_edge_aggr_w<<<cdiv((long long)ET * 32, 256), 256, 0, stream>>>(ei, Ee, Sm, Hb, Ob, NE, NN, 1, OUTC, NP2);
  k_logsoftmax<<<cdiv((long long)NN, 256), 256, 0, stream>>>(Ob, y, NN, OUTC, NP2);
}
